// MultiheadAttention_84413287236267
// MI455X (gfx1250) — matrix operand order repaired and checked by reading, not compiled, not run on hardware
//
#include <hip/hip_runtime.h>
#include <hip/hip_bf16.h>
#include <math.h>

// ---------------------------------------------------------------------------
// Shapes (fixed by the reference)
// ---------------------------------------------------------------------------
#define EMBED   768
#define NHEAD   12
#define HDIM    64
#define GRP     64
#define BATCH   4
#define SEQ     4096
#define MROWS   (BATCH * SEQ)        // 16384 token rows
#define NQKV    (3 * EMBED)          // 2304 fused qkv output columns
#define KDIM    768                  // shared K for both GEMMs

typedef __attribute__((ext_vector_type(16))) __bf16 v16bf;
typedef __attribute__((ext_vector_type(8)))  __bf16 v8bf;
typedef __attribute__((ext_vector_type(8)))  float  v8f;

// Global (address_space(1)) pointer types; constructed via inttoptr so the
// loads/stores lower to global_load_* / global_store_* (not flat_*).
typedef const v8bf  __attribute__((address_space(1)))* gc_v8bf_p;
typedef float       __attribute__((address_space(1)))* g_f32_p;
typedef __bf16      __attribute__((address_space(1)))* g_bf16_p;
typedef v8bf        __attribute__((address_space(1)))* g_v8bf_p;

#define AS1C(p) ((gc_v8bf_p)(size_t)(const void*)(p))

// ---------------------------------------------------------------------------
// Conversion kernels: f32 -> bf16, and weight transposes so that the WMMA
// B-fragment (lane = output column, contiguous K run) is two b128 loads.
// ---------------------------------------------------------------------------
__global__ __launch_bounds__(256) void cvt_f32_to_bf16(const float* __restrict__ src,
                                                       __bf16* __restrict__ dst, int n) {
    int i = blockIdx.x * blockDim.x + threadIdx.x;
    if (i < n) dst[i] = (__bf16)src[i];
}

// Wt[n][k] = concat(wq,wk,wv)[k][n mod 768], row-major [2304][768]
__global__ __launch_bounds__(256) void build_wqkv_t(const float* __restrict__ wq,
                                                    const float* __restrict__ wk,
                                                    const float* __restrict__ wv,
                                                    __bf16* __restrict__ wt) {
    int idx = blockIdx.x * blockDim.x + threadIdx.x;   // idx = n*768 + k
    if (idx >= NQKV * EMBED) return;
    int n = idx / EMBED;
    int k = idx - n * EMBED;
    float v;
    if (n < EMBED)            v = wq[k * EMBED + n];
    else if (n < 2 * EMBED)   v = wk[k * EMBED + (n - EMBED)];
    else                      v = wv[k * EMBED + (n - 2 * EMBED)];
    wt[idx] = (__bf16)v;
}

// Wt[n][k] = wo[k][n], row-major [768][768]
__global__ __launch_bounds__(256) void build_wo_t(const float* __restrict__ wo,
                                                  __bf16* __restrict__ wt) {
    int idx = blockIdx.x * blockDim.x + threadIdx.x;
    if (idx >= EMBED * EMBED) return;
    int n = idx / EMBED;
    int k = idx - n * EMBED;
    wt[idx] = (__bf16)wo[k * EMBED + n];
}

// ---------------------------------------------------------------------------
// WMMA bf16 GEMM:  C[M][N] = A[M][K=768] * B^T[N][K=768]  (B transposed)
// One wave -> 64x64 output tile (4x4 of 16x16 WMMA tiles => 16 WMMAs and
// 16 global_load_b128 per K-step, 1:1 wmma:load). 8 waves/block laid out
// 4 (M) x 2 (N) -> 256x128 per block. K stepped by 32. Fragment rows are
// compile-time byte offsets (i*16*768*2 = 24576 B) so loads use IOFFSET.
//
// Fragment addressing per CDNA5 ISA 7.12.2 (wave32):
//   A (16-bit, 16x32):  lane = (kg<<4)|m. VGPR0..3: K = kg*8+0..7 ;
//     VGPR4..7: K = 16+kg*8+0..7  -> two contiguous 16B runs per lane.
//   B (16-bit, 32x16):  lane = (kg<<4)|n. VGPR0..3: K = kg*8+0..7 ; VGPR4..7: K = 16+kg*8+0..7 ->
//     two contiguous 16B runs per lane (same as A) when B is stored N-major.
//   C/D (f32 16x16):    lanes 0..15: N=lane, VGPR r: M=r ; lanes 16..31: M=8+r.
// ---------------------------------------------------------------------------
template <bool OUT_F32>
__global__ __launch_bounds__(256) void gemm_bf16_wmma(const __bf16* __restrict__ A,
                                                      const __bf16* __restrict__ Bt,
                                                      void* __restrict__ C,
                                                      int M, int N) {
    const int lane = threadIdx.x & 31;
    const int wave = threadIdx.x >> 5;
    const int wm   = wave & 3;           // 4 waves down M
    const int wn   = wave >> 2;          // 2 waves across N
    const int mbase = blockIdx.y * 256 + wm * 64;
    const int nbase = blockIdx.x * 128 + wn * 64;
    const int lrow  = lane & 15;
    const int kg    = lane >> 4;

    // Single base address per matrix; fragment rows at constant offsets.
    const __bf16* aBase = A  + (size_t)(mbase + lrow) * KDIM + kg * 8;
    const __bf16* bBase = Bt + (size_t)(nbase + lrow) * KDIM + kg * 8;

    v8f acc[4][4];
#pragma unroll
    for (int i = 0; i < 4; ++i)
#pragma unroll
        for (int j = 0; j < 4; ++j) acc[i][j] = v8f{};

#pragma unroll 1
    for (int k0 = 0; k0 < KDIM; k0 += 32) {
        v16bf afrag[4], bfrag[4];
#pragma unroll
        for (int i = 0; i < 4; ++i) {
            gc_v8bf_p pa = AS1C(aBase + i * (16 * KDIM) + k0);
            ((v8bf*)&afrag[i])[0] = pa[0];   // K = kg*8 .. +7
            ((v8bf*)&afrag[i])[1] = pa[2];   // K = 16+kg*8 .. +7   (+32 B)
            gc_v8bf_p pb = AS1C(bBase + i * (16 * KDIM) + k0);
            ((v8bf*)&bfrag[i])[0] = pb[0];   // K = kg*8 .. +7
            ((v8bf*)&bfrag[i])[1] = pb[2];   // K = 16+kg*8 .. +7   (+32 B)
        }
        if (k0 + 64 < KDIM) {
#pragma unroll
            for (int i = 0; i < 4; ++i) {
                __builtin_prefetch(aBase + i * (16 * KDIM) + k0 + 64, 0, 1);
                __builtin_prefetch(bBase + i * (16 * KDIM) + k0 + 64, 0, 1);
            }
        }
#pragma unroll
        for (int i = 0; i < 4; ++i)
#pragma unroll
            for (int j = 0; j < 4; ++j)
                acc[i][j] = __builtin_amdgcn_wmma_f32_16x16x32_bf16(
                    false, afrag[i], false, bfrag[j],
                    (short)0, acc[i][j], false, false);
    }

#pragma unroll
    for (int i = 0; i < 4; ++i)
#pragma unroll
        for (int j = 0; j < 4; ++j) {
            int col = nbase + j * 16 + lrow;
#pragma unroll
            for (int r = 0; r < 8; ++r) {
                int row = mbase + i * 16 + kg * 8 + r;
                if (OUT_F32) {
                    g_f32_p dst = (g_f32_p)(size_t)((float*)C + (size_t)row * N + col);
                    *dst = acc[i][j][r];
                } else {
                    g_bf16_p dst = (g_bf16_p)(size_t)((__bf16*)C + (size_t)row * N + col);
                    *dst = (__bf16)acc[i][j][r];
                }
            }
        }
}

// ---------------------------------------------------------------------------
// Middle stage: RoPE(q,k) -> per-position dot -> softmax over 64-pos groups
// -> prob * v.  One 64-thread block per (b, h, group); thread = position.
// All q/k/v traffic is 16B-vectorized global_load/store_b128.
// ---------------------------------------------------------------------------
__global__ __launch_bounds__(64) void attn_group_softmax(const __bf16* __restrict__ QKV,
                                                         __bf16* __restrict__ Attn) {
    const int t = threadIdx.x;                 // position within group
    int cell = blockIdx.x;
    const int h   = cell % NHEAD; cell /= NHEAD;
    const int grp = cell % (SEQ / GRP);
    const int b   = cell / (SEQ / GRP);
    const int pos = grp * GRP + t;             // position within sequence
    const size_t m = (size_t)b * SEQ + pos;    // token row

    gc_v8bf_p qp = AS1C(QKV + m * NQKV + h * HDIM);
    gc_v8bf_p kp = AS1C(QKV + m * NQKV + EMBED + h * HDIM);
    gc_v8bf_p vp = AS1C(QKV + m * NQKV + 2 * EMBED + h * HDIM);

    // RoPE + dot (same-position dot: rotation cancels analytically, but we
    // compute it faithfully per the reference).
    const float lstep = 9.210340371976184f / 32.0f;   // ln(10000)/half
    float score = 0.0f;
#pragma unroll
    for (int c = 0; c < 8; ++c) {              // 8 elems = 4 rotary pairs each
        v8bf qc = qp[c];
        v8bf kc = kp[c];
#pragma unroll
        for (int p = 0; p < 4; ++p) {
            int j = c * 4 + p;                 // rotary pair index 0..31
            float invf = __expf(-(float)j * lstep);
            float ang  = (float)pos * invf;
            float cs = __cosf(ang), sn = __sinf(ang);
            float q1 = (float)qc[2 * p], q2 = (float)qc[2 * p + 1];
            float k1 = (float)kc[2 * p], k2 = (float)kc[2 * p + 1];
            float qr1 = q1 * cs - q2 * sn, qr2 = q1 * sn + q2 * cs;
            float kr1 = k1 * cs - k2 * sn, kr2 = k1 * sn + k2 * cs;
            score += qr1 * kr1 + qr2 * kr2;
        }
    }
    score *= 0.125f;                           // 1/sqrt(64)

    __shared__ float sm[GRP];
    sm[t] = score;
    __syncthreads();

    float mx = -3.402823466e38f;
#pragma unroll 8
    for (int i = 0; i < GRP; ++i) mx = fmaxf(mx, sm[i]);
    float sum = 0.0f;
#pragma unroll 8
    for (int i = 0; i < GRP; ++i) sum += __expf(sm[i] - mx);
    const float prob = __expf(score - mx) / sum;

    g_v8bf_p op = (g_v8bf_p)(size_t)(Attn + m * EMBED + h * HDIM);
#pragma unroll
    for (int c = 0; c < 8; ++c) {
        v8bf vc = vp[c];
        v8bf oc;
#pragma unroll
        for (int e = 0; e < 8; ++e)
            oc[e] = (__bf16)(prob * (float)vc[e]);
        op[c] = oc;                            // 16B global store
    }
}

// ---------------------------------------------------------------------------
// Launch
// ---------------------------------------------------------------------------
extern "C" void kernel_launch(void* const* d_in, const int* in_sizes, int n_in,
                              void* d_out, int out_size, void* d_ws, size_t ws_size,
                              hipStream_t stream) {
    const float* x  = (const float*)d_in[0];
    const float* wq = (const float*)d_in[1];
    const float* wk = (const float*)d_in[2];
    const float* wv = (const float*)d_in[3];
    const float* wo = (const float*)d_in[4];
    float* out = (float*)d_out;

    // Workspace layout (bf16 staging, ~130 MB total)
    char* ws = (char*)d_ws;
    size_t off = 0;
    __bf16* Xb    = (__bf16*)(ws + off); off += (size_t)MROWS * EMBED * 2;   // 24 MB
    __bf16* Wqkvt = (__bf16*)(ws + off); off += (size_t)NQKV  * EMBED * 2;   // 3.4 MB
    __bf16* Wot   = (__bf16*)(ws + off); off += (size_t)EMBED * EMBED * 2;   // 1.1 MB
    __bf16* QKV   = (__bf16*)(ws + off); off += (size_t)MROWS * NQKV  * 2;   // 75 MB
    __bf16* Attn  = (__bf16*)(ws + off); off += (size_t)MROWS * EMBED * 2;   // 24 MB
    (void)ws_size; (void)in_sizes; (void)n_in; (void)out_size;

    // 1) fp32 -> bf16 staging
    {
        int n = MROWS * EMBED;
        cvt_f32_to_bf16<<<(n + 255) / 256, 256, 0, stream>>>(x, Xb, n);
        int nw = NQKV * EMBED;
        build_wqkv_t<<<(nw + 255) / 256, 256, 0, stream>>>(wq, wk, wv, Wqkvt);
        int no = EMBED * EMBED;
        build_wo_t<<<(no + 255) / 256, 256, 0, stream>>>(wo, Wot);
    }

    // 2) fused QKV projection: [16384 x 768] @ [768 x 2304] -> bf16 QKV
    {
        dim3 grid(NQKV / 128, MROWS / 256);   // 18 x 64
        gemm_bf16_wmma<false><<<grid, 256, 0, stream>>>(Xb, Wqkvt, (void*)QKV,
                                                        MROWS, NQKV);
    }

    // 3) RoPE + grouped-softmax gating
    {
        int cells = BATCH * (SEQ / GRP) * NHEAD;  // 3072
        attn_group_softmax<<<cells, 64, 0, stream>>>(QKV, Attn);
    }

    // 4) output projection: [16384 x 768] @ [768 x 768] -> f32 d_out
    {
        dim3 grid(EMBED / 128, MROWS / 256);  // 6 x 64
        gemm_bf16_wmma<true><<<grid, 256, 0, stream>>>(Attn, Wot, (void*)out,
                                                       MROWS, EMBED);
    }
}
